// TrxGNNGPT_42296837931759
// MI455X (gfx1250) — matrix operand order repaired and checked by reading, not compiled, not run on hardware
//
#include <hip/hip_runtime.h>
#include <math.h>

// ---------------------------------------------------------------------------
// Problem constants (match reference)
// ---------------------------------------------------------------------------
#define V_ 32000
#define D_ 64
#define L_ 16
#define N_ 4096
#define E_ 8192
#define H_ 768
#define F_ 3072

typedef __bf16 bf16_t;
typedef __attribute__((ext_vector_type(8)))  __bf16 v8bf;
typedef __attribute__((ext_vector_type(16))) __bf16 v16bf;
typedef __attribute__((ext_vector_type(8)))  float  v8f;

// ---------------------------------------------------------------------------
// Small helpers
// ---------------------------------------------------------------------------
__device__ __forceinline__ v16bf ld16(const bf16_t* lo, const bf16_t* hi) {
  v8bf l = *(const v8bf*)lo;
  v8bf h = *(const v8bf*)hi;
  return __builtin_shufflevector(l, h, 0,1,2,3,4,5,6,7,8,9,10,11,12,13,14,15);
}

#define WMMA_BF16(a, b, c)                                                    \
  __builtin_amdgcn_wmma_f32_16x16x32_bf16(false, (a), false, (b), (short)0,   \
                                          (c), false, false)

// ---------------------------------------------------------------------------
// Generic WMMA GEMM:  D = act( A[M,K](bf16) @ B[K,N] + bias + res )
// B supplied transposed: BT[N,K] row-major (bf16).
// Wave tile 32x32 (2x2 fragments); block = 8 waves -> 128(M) x 64(N).
// Requires M%128==0, N%64==0, K%32==0.
// ---------------------------------------------------------------------------
__device__ __forceinline__ void epilogue(
    v8f c, int mbase, int nbase, int lane, int N,
    const float* __restrict__ bias, const float* __restrict__ res,
    float* __restrict__ Df, bf16_t* __restrict__ Db, int act)
{
  const int nn   = nbase + (lane & 15);
  const int mrow = mbase + (lane >> 4) * 8;
  const float bv = bias ? bias[nn] : 0.0f;
#pragma unroll
  for (int v = 0; v < 8; ++v) {
    float val = c[v] + bv;
    const size_t idx = (size_t)(mrow + v) * (size_t)N + (size_t)nn;
    if (res) val += res[idx];
    if (act == 1) {
      val = fmaxf(val, 0.0f);
    } else if (act == 2) {
      const float u = 0.7978845608028654f * (val + 0.044715f * val * val * val);
      val = 0.5f * val * (1.0f + tanhf(u));
    }
    if (Df) Df[idx] = val;
    if (Db) Db[idx] = (bf16_t)val;
  }
}

__global__ __launch_bounds__(256) void k_wmma_gemm(
    const bf16_t* __restrict__ A, const bf16_t* __restrict__ BT,
    const float* __restrict__ bias, const float* __restrict__ res,
    float* __restrict__ Df, bf16_t* __restrict__ Db,
    int M, int N, int K, int act)
{
  const int lane = threadIdx.x & 31;
  const int wave = threadIdx.x >> 5;
  const int m0 = blockIdx.y * 128 + (wave & 3) * 32;
  const int n0 = blockIdx.x * 64  + (wave >> 2) * 32;
  const int r  = lane & 15;
  const int hs = lane >> 4;

  const bf16_t* a0p = A  + (size_t)(m0 + r) * (size_t)K + (size_t)(hs * 8);
  const bf16_t* a1p = a0p + (size_t)16 * (size_t)K;
  const bf16_t* b0p = BT + (size_t)(n0 + r) * (size_t)K + (size_t)(hs * 8);
  const bf16_t* b1p = b0p + (size_t)16 * (size_t)K;

  v8f c00 = {}, c01 = {}, c10 = {}, c11 = {};
  for (int k = 0; k < K; k += 32) {
    __builtin_prefetch(a0p + k + 128, 0, 0);
    __builtin_prefetch(b0p + k + 128, 0, 0);
    v16bf a0 = ld16(a0p + k, a0p + k + 16);
    v16bf a1 = ld16(a1p + k, a1p + k + 16);
    v16bf b0 = ld16(b0p + k, b0p + k + 16);
    v16bf b1 = ld16(b1p + k, b1p + k + 16);
    c00 = WMMA_BF16(a0, b0, c00);
    c01 = WMMA_BF16(a0, b1, c01);
    c10 = WMMA_BF16(a1, b0, c10);
    c11 = WMMA_BF16(a1, b1, c11);
  }
  epilogue(c00, m0,      n0,      lane, N, bias, res, Df, Db, act);
  epilogue(c01, m0,      n0 + 16, lane, N, bias, res, Df, Db, act);
  epilogue(c10, m0 + 16, n0,      lane, N, bias, res, Df, Db, act);
  epilogue(c11, m0 + 16, n0 + 16, lane, N, bias, res, Df, Db, act);
}

// ---------------------------------------------------------------------------
// Fused tied-LM-head + softmax:
//   probs[m, :] = softmax(proj[m, :] @ emb^T)  written directly, one pass of
//   HBM traffic. Block = 16 rows, 8 waves sweep V in interleaved 16-col tiles.
//   Pass A: WMMA logits tiles -> online (max, sumexp) in registers.
//   Pass B: recompute tiles (L2 hits) -> write exp(x - M) / S.
// ---------------------------------------------------------------------------
__global__ __launch_bounds__(256) void k_lmhead_softmax(
    const bf16_t* __restrict__ proj,   // [E_, 64]
    const bf16_t* __restrict__ embb,   // [V_, 64]  == B^T for tied head
    float* __restrict__ probs)         // [E_, V_]
{
  __shared__ float red_m[8][16];
  __shared__ float red_s[8][16];
  __shared__ float fin_m[16];
  __shared__ float fin_is[16];

  const int lane = threadIdx.x & 31;
  const int wave = threadIdx.x >> 5;
  const int m0 = blockIdx.x * 16;
  const int r  = lane & 15;
  const int hs = lane >> 4;

  // A fragments: proj rows m0..m0+15, K = 64 -> two 32-wide K chunks.
  const bf16_t* ap = proj + (size_t)(m0 + r) * D_ + (size_t)(hs * 8);
  const v16bf a0 = ld16(ap,      ap + 16);
  const v16bf a1 = ld16(ap + 32, ap + 48);

  float m[8], s[8];
#pragma unroll
  for (int v = 0; v < 8; ++v) { m[v] = -3.0e38f; s[v] = 0.0f; }

  // ---- pass A: online softmax statistics ----------------------------------
  for (int t = wave; t < V_ / 16; t += 8) {
    const bf16_t* bp = embb + (size_t)(t * 16 + r) * D_ + (size_t)(hs * 8);
    v16bf b0 = ld16(bp,      bp + 16);
    v16bf b1 = ld16(bp + 32, bp + 48);
    v8f c = {};
    c = WMMA_BF16(a0, b0, c);
    c = WMMA_BF16(a1, b1, c);
#pragma unroll
    for (int v = 0; v < 8; ++v) {
      const float x = c[v];
      if (x > m[v]) { s[v] = s[v] * __expf(m[v] - x) + 1.0f; m[v] = x; }
      else          { s[v] += __expf(x - m[v]); }
    }
  }

  // lane butterfly within each 16-lane half (lanes sharing a row)
#pragma unroll
  for (int v = 0; v < 8; ++v) {
#pragma unroll
    for (int off = 1; off < 16; off <<= 1) {
      const float om = __shfl_xor(m[v], off, 32);
      const float os = __shfl_xor(s[v], off, 32);
      const float nm = fmaxf(m[v], om);
      s[v] = s[v] * __expf(m[v] - nm) + os * __expf(om - nm);
      m[v] = nm;
    }
  }
  if (r == 0) {
#pragma unroll
    for (int v = 0; v < 8; ++v) {
      red_m[wave][v + hs * 8] = m[v];
      red_s[wave][v + hs * 8] = s[v];
    }
  }
  __syncthreads();
  if (threadIdx.x < 16) {
    float M = -3.0e38f, S = 0.0f;
    for (int wv = 0; wv < 8; ++wv) {
      const float om = red_m[wv][threadIdx.x];
      const float os = red_s[wv][threadIdx.x];
      const float nm = fmaxf(M, om);
      S = S * __expf(M - nm) + os * __expf(om - nm);
      M = nm;
    }
    fin_m[threadIdx.x]  = M;
    fin_is[threadIdx.x] = 1.0f / S;
  }
  __syncthreads();

  float rowM[8], rowIS[8];
#pragma unroll
  for (int v = 0; v < 8; ++v) {
    rowM[v]  = fin_m[v + hs * 8];
    rowIS[v] = fin_is[v + hs * 8];
  }

  // ---- pass B: recompute tiles (bit-identical) and emit probabilities -----
  for (int t = wave; t < V_ / 16; t += 8) {
    const int n0 = t * 16;
    const bf16_t* bp = embb + (size_t)(n0 + r) * D_ + (size_t)(hs * 8);
    v16bf b0 = ld16(bp,      bp + 16);
    v16bf b1 = ld16(bp + 32, bp + 48);
    v8f c = {};
    c = WMMA_BF16(a0, b0, c);
    c = WMMA_BF16(a1, b1, c);
    const int nn = n0 + r;
#pragma unroll
    for (int v = 0; v < 8; ++v) {
      const int row = m0 + v + hs * 8;
      probs[(size_t)row * V_ + nn] = __expf(c[v] - rowM[v]) * rowIS[v];
    }
  }
}

// ---------------------------------------------------------------------------
// Elementwise / data-movement kernels
// ---------------------------------------------------------------------------
__global__ void k_cast_bf16(const float* __restrict__ s, bf16_t* __restrict__ d,
                            long n) {
  long i = blockIdx.x * (long)blockDim.x + threadIdx.x;
  if (i < n) d[i] = (bf16_t)s[i];
}

// src[R,C] (f32) -> dst[C,R] (bf16)
__global__ void k_transpose_bf16(const float* __restrict__ s,
                                 bf16_t* __restrict__ d, int R, int C) {
  long i = blockIdx.x * (long)blockDim.x + threadIdx.x;
  long total = (long)R * C;
  if (i < total) {
    int r = (int)(i / C);
    int c = (int)(i % C);
    d[(long)c * R + r] = (bf16_t)s[i];
  }
}

// out[row, k] = bf16(emb[toks[row, k/64], k%64]),  rows x (L*D)
__global__ void k_gather_embed(const int* __restrict__ toks,
                               const float* __restrict__ emb,
                               bf16_t* __restrict__ out, int rows) {
  long i = blockIdx.x * (long)blockDim.x + threadIdx.x;
  long total = (long)rows * (L_ * D_);
  if (i < total) {
    int row = (int)(i / (L_ * D_));
    int k   = (int)(i % (L_ * D_));
    int t   = toks[row * L_ + (k >> 6)];
    out[i]  = (bf16_t)emb[(long)t * D_ + (k & 63)];
  }
}

__global__ void k_zero_f32(float* __restrict__ p, long n) {
  long i = blockIdx.x * (long)blockDim.x + threadIdx.x;
  if (i < n) p[i] = 0.0f;
}

// msg = h[src] + tmpe; agg[dst] += msg  (atomic scatter-add)
__global__ void k_msg_scatter(const float* __restrict__ h,
                              const float* __restrict__ tmpe,
                              const int* __restrict__ ei,
                              float* __restrict__ agg) {
  long i = blockIdx.x * (long)blockDim.x + threadIdx.x;
  if (i < (long)E_ * D_) {
    int e = (int)(i >> 6);
    int d = (int)(i & 63);
    int s = ei[e];
    int t = ei[E_ + e];
    float m = h[(long)s * D_ + d] + tmpe[i];
    atomicAdd(&agg[(long)t * D_ + d], m);
  }
}

__global__ void k_add_cast(const float* __restrict__ a,
                           const float* __restrict__ b,
                           bf16_t* __restrict__ o, long n) {
  long i = blockIdx.x * (long)blockDim.x + threadIdx.x;
  if (i < n) o[i] = (bf16_t)(a[i] + b[i]);
}

// edge_emb = node_out[src] + node_out[dst]   (bf16 out)
__global__ void k_edge_emb(const float* __restrict__ nodeo,
                           const int* __restrict__ ei,
                           bf16_t* __restrict__ o) {
  long i = blockIdx.x * (long)blockDim.x + threadIdx.x;
  if (i < (long)E_ * D_) {
    int e = (int)(i >> 6);
    int d = (int)(i & 63);
    float v = nodeo[(long)ei[e] * D_ + d] + nodeo[(long)ei[E_ + e] * D_ + d];
    o[i] = (bf16_t)v;
  }
}

// LayerNorm over last dim Hn; one block per row; bf16 out
__global__ __launch_bounds__(256) void k_layernorm(
    const float* __restrict__ x, const float* __restrict__ g,
    const float* __restrict__ b, bf16_t* __restrict__ o, int Hn) {
  __shared__ float red[256];
  const int row = blockIdx.x;
  const float* xr = x + (size_t)row * Hn;

  float s = 0.0f;
  for (int i = threadIdx.x; i < Hn; i += 256) s += xr[i];
  red[threadIdx.x] = s; __syncthreads();
  for (int off = 128; off > 0; off >>= 1) {
    if ((int)threadIdx.x < off) red[threadIdx.x] += red[threadIdx.x + off];
    __syncthreads();
  }
  const float mu = red[0] / Hn;
  __syncthreads();

  float v = 0.0f;
  for (int i = threadIdx.x; i < Hn; i += 256) {
    float t = xr[i] - mu; v += t * t;
  }
  red[threadIdx.x] = v; __syncthreads();
  for (int off = 128; off > 0; off >>= 1) {
    if ((int)threadIdx.x < off) red[threadIdx.x] += red[threadIdx.x + off];
    __syncthreads();
  }
  const float rstd = rsqrtf(red[0] / Hn + 1e-5f);

  for (int i = threadIdx.x; i < Hn; i += 256)
    o[(size_t)row * Hn + i] = (bf16_t)((xr[i] - mu) * rstd * g[i] + b[i]);
}

__global__ void k_labels(const int* __restrict__ t, float* __restrict__ o,
                         long n) {
  long i = blockIdx.x * (long)blockDim.x + threadIdx.x;
  if (i < n) o[i] = (float)t[i];
}

// ---------------------------------------------------------------------------
// Host-side orchestration
// ---------------------------------------------------------------------------
extern "C" void kernel_launch(void* const* d_in, const int* in_sizes, int n_in,
                              void* d_out, int out_size, void* d_ws,
                              size_t ws_size, hipStream_t stream) {
  const int*   x_tokens    = (const int*)d_in[0];
  const int*   edge_tokens = (const int*)d_in[1];
  const int*   edge_index  = (const int*)d_in[2];
  const float* emb  = (const float*)d_in[3];
  const float* W1   = (const float*)d_in[4];
  const float* b1   = (const float*)d_in[5];
  const float* We   = (const float*)d_in[6];
  const float* W2   = (const float*)d_in[7];
  const float* b2   = (const float*)d_in[8];
  const float* Wc1  = (const float*)d_in[9];
  const float* bc1  = (const float*)d_in[10];
  const float* ln_g = (const float*)d_in[11];
  const float* ln_b = (const float*)d_in[12];
  const float* Wt1  = (const float*)d_in[13];
  const float* bt1  = (const float*)d_in[14];
  const float* Wt2  = (const float*)d_in[15];
  const float* bt2  = (const float*)d_in[16];
  const float* Wc2  = (const float*)d_in[17];
  const float* bc2  = (const float*)d_in[18];

  float* out = (float*)d_out;

  // ---- workspace carve-out -------------------------------------------------
  char* w = (char*)d_ws;
  auto alloc = [&](size_t bytes) -> void* {
    void* p = (void*)w;
    w += (bytes + 255) & ~(size_t)255;
    return p;
  };
  bf16_t* emb_bf      = (bf16_t*)alloc((size_t)V_ * D_ * 2);
  bf16_t* node_tok_bf = (bf16_t*)alloc((size_t)N_ * L_ * D_ * 2);
  bf16_t* edge_tok_bf = (bf16_t*)alloc((size_t)E_ * L_ * D_ * 2);
  bf16_t* W1T  = (bf16_t*)alloc((size_t)L_ * D_ * D_ * 2);
  bf16_t* WeT  = (bf16_t*)alloc((size_t)L_ * D_ * D_ * 2);
  bf16_t* W2T  = (bf16_t*)alloc((size_t)D_ * D_ * 2);
  bf16_t* Wc1T = (bf16_t*)alloc((size_t)D_ * H_ * 2);
  bf16_t* Wt1T = (bf16_t*)alloc((size_t)H_ * F_ * 2);
  bf16_t* Wt2T = (bf16_t*)alloc((size_t)F_ * H_ * 2);
  bf16_t* Wc2T = (bf16_t*)alloc((size_t)H_ * D_ * 2);
  float*  h        = (float*)alloc((size_t)N_ * D_ * 4);
  float*  tmpe     = (float*)alloc((size_t)E_ * D_ * 4);
  float*  agg      = (float*)alloc((size_t)N_ * D_ * 4);
  bf16_t* h2_bf    = (bf16_t*)alloc((size_t)N_ * D_ * 2);
  float*  node_out = (float*)alloc((size_t)N_ * D_ * 4);
  bf16_t* eemb_bf  = (bf16_t*)alloc((size_t)E_ * D_ * 2);
  float*  x_f      = (float*)alloc((size_t)E_ * H_ * 4);
  bf16_t* xn_bf    = (bf16_t*)alloc((size_t)E_ * H_ * 2);
  bf16_t* act_bf   = (bf16_t*)alloc((size_t)E_ * F_ * 2);
  bf16_t* hid_bf   = (bf16_t*)alloc((size_t)E_ * H_ * 2);
  bf16_t* proj_bf  = (bf16_t*)alloc((size_t)E_ * D_ * 2);
  (void)ws_size; (void)in_sizes; (void)n_in; (void)out_size;

  const dim3 B256(256);
  auto blocks = [](long n) { return dim3((unsigned)((n + 255) / 256)); };
  auto gemm = [&](const bf16_t* Ap, const bf16_t* Bp, const float* bi,
                  const float* rs, float* Df, bf16_t* Db, int M, int N, int K,
                  int act) {
    dim3 grid((unsigned)(N / 64), (unsigned)(M / 128));
    k_wmma_gemm<<<grid, B256, 0, stream>>>(Ap, Bp, bi, rs, Df, Db, M, N, K, act);
  };

  // ---- precision conversion + weight transposes ---------------------------
  k_cast_bf16<<<blocks((long)V_ * D_), B256, 0, stream>>>(emb, emb_bf, (long)V_ * D_);
  k_transpose_bf16<<<blocks((long)L_ * D_ * D_), B256, 0, stream>>>(W1,  W1T,  L_ * D_, D_);
  k_transpose_bf16<<<blocks((long)L_ * D_ * D_), B256, 0, stream>>>(We,  WeT,  L_ * D_, D_);
  k_transpose_bf16<<<blocks((long)D_ * D_),      B256, 0, stream>>>(W2,  W2T,  D_, D_);
  k_transpose_bf16<<<blocks((long)D_ * H_),      B256, 0, stream>>>(Wc1, Wc1T, D_, H_);
  k_transpose_bf16<<<blocks((long)H_ * F_),      B256, 0, stream>>>(Wt1, Wt1T, H_, F_);
  k_transpose_bf16<<<blocks((long)F_ * H_),      B256, 0, stream>>>(Wt2, Wt2T, F_, H_);
  k_transpose_bf16<<<blocks((long)H_ * D_),      B256, 0, stream>>>(Wc2, Wc2T, H_, D_);

  // ---- embedding gathers ---------------------------------------------------
  k_gather_embed<<<blocks((long)N_ * L_ * D_), B256, 0, stream>>>(x_tokens, emb, node_tok_bf, N_);
  k_gather_embed<<<blocks((long)E_ * L_ * D_), B256, 0, stream>>>(edge_tokens, emb, edge_tok_bf, E_);

  // ---- GNN layer -----------------------------------------------------------
  k_zero_f32<<<blocks((long)N_ * D_), B256, 0, stream>>>(agg, (long)N_ * D_);
  // h = relu(node_tok @ W1 + b1)                        [4096,1024]@[1024,64]
  gemm(node_tok_bf, W1T, b1, nullptr, h, nullptr, N_, D_, L_ * D_, 1);
  // tmpe = edge_tok @ We                                [8192,1024]@[1024,64]
  gemm(edge_tok_bf, WeT, nullptr, nullptr, tmpe, nullptr, E_, D_, L_ * D_, 0);
  // agg[dst] += h[src] + tmpe
  k_msg_scatter<<<blocks((long)E_ * D_), B256, 0, stream>>>(h, tmpe, edge_index, agg);
  // node_out = relu((h + agg) @ W2 + b2)
  k_add_cast<<<blocks((long)N_ * D_), B256, 0, stream>>>(h, agg, h2_bf, (long)N_ * D_);
  gemm(h2_bf, W2T, b2, nullptr, node_out, nullptr, N_, D_, D_, 1);

  // ---- edge aggregation + transformer block -------------------------------
  k_edge_emb<<<blocks((long)E_ * D_), B256, 0, stream>>>(node_out, edge_index, eemb_bf);
  // x = edge_emb @ Wc1 + bc1                             [8192,64]@[64,768]
  gemm(eemb_bf, Wc1T, bc1, nullptr, x_f, nullptr, E_, H_, D_, 0);
  // xn = layernorm(x) * g + b
  k_layernorm<<<dim3(E_), B256, 0, stream>>>(x_f, ln_g, ln_b, xn_bf, H_);
  // act = gelu(xn @ Wt1 + bt1)                           [8192,768]@[768,3072]
  gemm(xn_bf, Wt1T, bt1, nullptr, nullptr, act_bf, E_, F_, H_, 2);
  // hidden = act @ Wt2 + bt2 + x                         [8192,3072]@[3072,768]
  gemm(act_bf, Wt2T, bt2, x_f, nullptr, hid_bf, E_, H_, F_, 0);
  // proj = hidden @ Wc2 + bc2                            [8192,768]@[768,64]
  gemm(hid_bf, Wc2T, bc2, nullptr, nullptr, proj_bf, E_, D_, H_, 0);

  // ---- fused tied LM head + softmax straight into d_out -------------------
  float* probs = out + (size_t)E_ * L_;  // after labels
  k_lmhead_softmax<<<dim3(E_ / 16), B256, 0, stream>>>(proj_bf, emb_bf, probs);

  // ---- labels = edge_tokens (as floats, output slot 0) --------------------
  k_labels<<<blocks((long)E_ * L_), B256, 0, stream>>>(edge_tokens, out, (long)E_ * L_);
}
